// Attention_41420664603305
// MI455X (gfx1250) — matrix operand order repaired and checked by reading, not compiled, not run on hardware
//
#include <hip/hip_runtime.h>
#include <hip/hip_bf16.h>

// MI455X / gfx1250, wave32. Fused MHA forward:
//   q,k,v = x@W{q,k,v}^T + b ; attn = softmax(q k^T) v ; out = x + attn
// bf16 WMMA everywhere; QKV GEMM stages the shared 64-row W strip into LDS
// with the Tensor Data Mover (double-buffered, TENSORcnt-synced), using TDM's
// pad feature (16B per 256B row) to spread LDS banks. v4: batch LDS B-tile
// loads ahead of the WMMA group so waits are partial instead of dscnt==0.

typedef __attribute__((ext_vector_type(16))) __bf16 v16bf;
typedef __attribute__((ext_vector_type(8)))  __bf16 v8bf;
typedef __attribute__((ext_vector_type(8)))  float  v8f;
typedef __attribute__((ext_vector_type(4)))  unsigned int v4u;
typedef __attribute__((ext_vector_type(8)))  int v8i;
typedef __attribute__((ext_vector_type(4)))  int v4i;

#define B_  4
#define N_  2048
#define D_  1024
#define H_  16
#define HD_ 64
#define M_  (B_*N_)          // 8192 tokens

// workspace layout (bf16 element offsets)
#define OFF_X  ((size_t)0)
#define OFF_W  ((size_t)M_*D_)                    // 3 weight matrices, D_*D_ each
#define OFF_Q  (OFF_W + (size_t)3*D_*D_)
#define OFF_K  (OFF_Q + (size_t)M_*D_)
#define OFF_V  (OFF_K + (size_t)M_*D_)
// total = 36.7M bf16 elems = ~73 MB of d_ws

#define KC_     128                 // k-chunk staged per TDM transfer
#define RSTRIDE 136                 // 128 elems + 8 elems (16B) TDM row pad

// ---------------------------------------------------------------------------
// Kernel 0: one-shot f32 -> bf16 conversion of X and Wq/Wk/Wv.
// ---------------------------------------------------------------------------
__global__ __launch_bounds__(256) void cvt_bf16(
    const float* __restrict__ x,
    const float* __restrict__ Wq,
    const float* __restrict__ Wk,
    const float* __restrict__ Wv,
    void* __restrict__ wsv)
{
  __bf16* ws = (__bf16*)wsv;
  const size_t t   = (size_t)blockIdx.x * blockDim.x + threadIdx.x;
  const size_t idx = t * 8;
  const size_t nx  = (size_t)M_ * D_;
  const size_t nw  = (size_t)D_ * D_;

  const float* src;
  if (idx < nx) {
    src = x + idx;
  } else {
    const size_t widx  = idx - nx;
    const int    which = (int)(widx / nw);
    const float* W     = (which == 0) ? Wq : (which == 1) ? Wk : Wv;
    src = W + (widx % nw);
  }
  v8f f = *(const v8f*)src;
  v8bf o;
  #pragma unroll
  for (int i = 0; i < 8; ++i) o[i] = (__bf16)f[i];
  *(v8bf*)(ws + idx) = o;
}

// ---------------------------------------------------------------------------
// TDM: async-load a 64-row x 128-col bf16 tile of W into LDS.
// D# group0/1 packed per CDNA5 ISA §8.3/§8.4:
//   g0: count=1 | lds_addr | global_addr | type=2
//   g1: data_size=2B, pad_enable, pad_interval=64 DWORDs, pad_amount=4 DWORDs,
//       tensor_dim0=128, tensor_dim1=64, tile_dim0=128, tile_dim1=64,
//       tensor_dim0_stride=1024
// Result rows land at stride 256+16 bytes (RSTRIDE elems).
// This toolchain exposes the 6-arg builtin: (g0, g1, g2, g3, g4, cpol).
// ---------------------------------------------------------------------------
__device__ __forceinline__ void tdm_load_w_tile(const __bf16* gsrc, unsigned lds_addr)
{
  const unsigned long long ga = (unsigned long long)(uintptr_t)gsrc;
  v4u g0;
  g0[0] = 1u;                                   // count=1 (valid descriptor)
  g0[1] = lds_addr;                             // LDS byte address
  g0[2] = (unsigned)ga;                         // global_addr[31:0]
  g0[3] = (unsigned)(ga >> 32) | (2u << 30);    // global_addr[56:32] | type=2
  v8i g1;
  g1[0] = (int)((1u << 16) | (1u << 20) | (5u << 22) | (3u << 25));
  g1[1] = (int)(128u << 16);                    // tensor_dim0 = 128
  g1[2] = (int)(64u  << 16);                    // tensor_dim1 = 64
  g1[3] = (int)(128u << 16);                    // tile_dim0   = 128
  g1[4] = (int)64u;                             // tile_dim1 = 64, tile_dim2 = 0
  g1[5] = (int)1024u;                           // tensor_dim0_stride = 1024
  g1[6] = 0;
  g1[7] = 0;
  v4i z4 = {0, 0, 0, 0};
  v8i z8 = {0, 0, 0, 0, 0, 0, 0, 0};
  __builtin_amdgcn_tensor_load_to_lds(g0, g1, z4, z4, z8, 0);
}

// ---------------------------------------------------------------------------
// Kernel 1: fused QKV projection GEMM  P = X @ W^T + b, all-bf16.
// One block = 128x64 output tile: 8 waves x 16 rows share a 64-col W strip
// staged through LDS by the TDM (double-buffered).
// Outputs:  Q : (B*H, N, HD)   K : (B*H, N, HD)   V^T : (B*H, HD, N)
// ---------------------------------------------------------------------------
__global__ __launch_bounds__(256) void qkv_gemm(
    const float* __restrict__ bq,
    const float* __restrict__ bk,
    const float* __restrict__ bv,
    void* __restrict__ wsv)
{
  __shared__ __align__(16) __bf16 ldsW[2][64 * RSTRIDE];   // 2 x 17 KB

  __bf16* ws = (__bf16*)wsv;
  const int lane = threadIdx.x & 31;
  const int wv   = threadIdx.x >> 5;
  const int blk  = blockIdx.x;                   // 0 .. 3*64*16-1
  const int which = blk / (64 * 16);             // 0=Q 1=K 2=V
  const int rem   = blk % (64 * 16);
  const int mtb   = rem / 16;                    // 128-row block
  const int nt    = rem % 16;                    // 64-col strip

  const float* bias = (which == 0) ? bq : (which == 1) ? bk : bv;

  const int m  = lane & 15;                      // A row / B col / C col
  const int hi = lane >> 4;                      // lane half
  const int c0 = hi * 8;                         // A-layout K chunk base
  const int kh = hi * 16;                        // B-layout K half base

  const __bf16* X  = ws + OFF_X;
  const __bf16* Wb = ws + OFF_W + (size_t)which * D_ * D_
                   + (size_t)(nt * 64) * D_;     // strip base (64 rows)

  const __bf16* xrow = X + (size_t)(mtb * 128 + wv * 16 + m) * D_;

  v8f acc[4] = {};

  // prologue: stage k-chunk 0
  if (wv == 0)
    tdm_load_w_tile(Wb, (unsigned)(uintptr_t)&ldsW[0][0]);

  for (int kc = 0; kc < D_ / KC_; ++kc) {
    const int buf = kc & 1;
    if (wv == 0) __builtin_amdgcn_s_wait_tensorcnt(0);   // tile ready
    __syncthreads();
    if (wv == 0 && kc + 1 < D_ / KC_)                    // prefetch next chunk
      tdm_load_w_tile(Wb + (size_t)(kc + 1) * KC_,
                      (unsigned)(uintptr_t)&ldsW[buf ^ 1][0]);

    #pragma unroll
    for (int kk = 0; kk < KC_ / 32; ++kk) {
      const int kg = kc * KC_ + kk * 32;
      // A operand: X 16x32 tile from global (per-lane two 8-elem chunks)
      v8bf a0 = *(const v8bf*)(xrow + kg + kh);
      v8bf a1 = *(const v8bf*)(xrow + kg + kh + 8);
      v16bf a;
      #pragma unroll
      for (int i = 0; i < 8; ++i) { a[i] = a0[i]; a[8 + i] = a1[i]; }
      // B operands: batch all 4 LDS tile reads first (one ds clause, partial
      // dscnt waits), then issue the 4 WMMAs back-to-back.
      v16bf b[4];
      #pragma unroll
      for (int c = 0; c < 4; ++c)
        b[c] = *(const v16bf*)&ldsW[buf][(c * 16 + m) * RSTRIDE + kk * 32 + kh];
      #pragma unroll
      for (int c = 0; c < 4; ++c)
        acc[c] = __builtin_amdgcn_wmma_f32_16x16x32_bf16(false, a, false, b[c],
                                                         (short)0, acc[c],
                                                         false, false);
    }
    __syncthreads();                              // all reads done before reuse
  }

  __bf16* Qb = ws + OFF_Q;
  __bf16* Kb = ws + OFF_K;
  __bf16* Vb = ws + OFF_V;
  #pragma unroll
  for (int c = 0; c < 4; ++c) {
    const int  ng    = nt * 64 + c * 16 + m;      // global feature col
    const float bcol = bias[ng];
    const int  head  = ng / HD_, dd = ng % HD_;
    #pragma unroll
    for (int r = 0; r < 8; ++r) {
      const int row  = r + hi * 8;                // C layout: M = r + 8*(lane>=16)
      const int mg   = mtb * 128 + wv * 16 + row;
      const int bidx = mg / N_, seq = mg % N_;
      const __bf16 bf = (__bf16)(acc[c][r] + bcol);
      if (which == 2) {                           // V stored transposed
        Vb[(((size_t)bidx * H_ + head) * HD_ + dd) * N_ + seq] = bf;
      } else {
        __bf16* P = (which == 0) ? Qb : Kb;
        P[(((size_t)bidx * H_ + head) * N_ + seq) * HD_ + dd] = bf;
      }
    }
  }
}

// ---------------------------------------------------------------------------
// Kernel 2: flash-attention. One wave owns a 16-query tile of one (b,head);
// streams keys/values in blocks of 32 with online softmax. O accum = 4x v8f.
// ---------------------------------------------------------------------------
__global__ __launch_bounds__(256) void attn(
    const float* __restrict__ x,
    const void* __restrict__ wsv,
    float* __restrict__ out)
{
  __shared__ __align__(32) float lds[8][16][32];   // per-wave P tile (C->A relayout)

  const __bf16* ws = (const __bf16*)wsv;
  const int lane = threadIdx.x & 31;
  const int wv   = threadIdx.x >> 5;
  const int gid  = blockIdx.x * 8 + wv;            // 0..8191
  const int qt   = gid & 127;                      // query tile within (b,h)
  const int bh   = gid >> 7;                       // 0..63

  const __bf16* Q  = ws + OFF_Q;
  const __bf16* K  = ws + OFF_K;
  const __bf16* Vt = ws + OFF_V;

  const int m  = lane & 15;
  const int hi = lane >> 4;
  const int c0 = hi * 8;
  const int kh = hi * 16;

  // Q tile (16 x 64) in A layout: two 16x32 chunks, kept in registers.
  const __bf16* qrow = Q + ((size_t)bh * N_ + qt * 16 + m) * HD_;
  v16bf aq0, aq1;
  {
    v8bf t0 = *(const v8bf*)(qrow + kh);
    v8bf t1 = *(const v8bf*)(qrow + kh + 8);
    v8bf t2 = *(const v8bf*)(qrow + 32 + kh);
    v8bf t3 = *(const v8bf*)(qrow + 32 + kh + 8);
    #pragma unroll
    for (int i = 0; i < 8; ++i) {
      aq0[i] = t0[i]; aq0[8 + i] = t1[i];
      aq1[i] = t2[i]; aq1[8 + i] = t3[i];
    }
  }

  float mrow[8], lrow[8];
  v8f acc[4] = {};
  #pragma unroll
  for (int r = 0; r < 8; ++r) { mrow[r] = -__builtin_inff(); lrow[r] = 0.f; }

  const __bf16* Kbh  = K  + (size_t)bh * N_ * HD_;
  const __bf16* Vtbh = Vt + (size_t)bh * HD_ * N_;

  for (int j = 0; j < N_; j += 32) {
    // prefetch next key block (global_prefetch_b8)
    if (j + 32 < N_) {
      __builtin_prefetch(Kbh + (size_t)(j + 32 + m) * HD_, 0, 1);
      __builtin_prefetch(Kbh + (size_t)(j + 48 + m) * HD_, 0, 1);
    }

    // ---- S = Q K^T for 2 key subtiles of 16 (contraction hd=64 -> 2 WMMAs) ----
    v8f S0 = {}, S1 = {};
    {
      const __bf16* kr0 = Kbh + (size_t)(j + m) * HD_;        // B col = key row
      const __bf16* kr1 = Kbh + (size_t)(j + 16 + m) * HD_;
      v16bf b00 = *(const v16bf*)(kr0 + kh);
      v16bf b01 = *(const v16bf*)(kr0 + 32 + kh);
      v16bf b10 = *(const v16bf*)(kr1 + kh);
      v16bf b11 = *(const v16bf*)(kr1 + 32 + kh);
      S0 = __builtin_amdgcn_wmma_f32_16x16x32_bf16(false, aq0, false, b00, (short)0, S0, false, false);
      S0 = __builtin_amdgcn_wmma_f32_16x16x32_bf16(false, aq1, false, b01, (short)0, S0, false, false);
      S1 = __builtin_amdgcn_wmma_f32_16x16x32_bf16(false, aq0, false, b10, (short)0, S1, false, false);
      S1 = __builtin_amdgcn_wmma_f32_16x16x32_bf16(false, aq1, false, b11, (short)0, S1, false, false);
    }

    // ---- online softmax: 16-lane groups == C-layout row split on wave32 ----
    float sc[8];
    #pragma unroll
    for (int r = 0; r < 8; ++r) {
      float mx = fmaxf(S0[r], S1[r]);
      #pragma unroll
      for (int off = 1; off < 16; off <<= 1) mx = fmaxf(mx, __shfl_xor(mx, off, 16));
      const float mn = fmaxf(mrow[r], mx);
      const float s  = __expf(mrow[r] - mn);
      const float p0 = __expf(S0[r] - mn);
      const float p1 = __expf(S1[r] - mn);
      float rs = p0 + p1;
      #pragma unroll
      for (int off = 1; off < 16; off <<= 1) rs += __shfl_xor(rs, off, 16);
      lrow[r] = lrow[r] * s + rs;
      mrow[r] = mn;
      sc[r]   = s;
      const int row = r + hi * 8;                  // C layout row
      lds[wv][row][m]      = p0;                   // P -> LDS for relayout
      lds[wv][row][16 + m] = p1;
    }
    #pragma unroll
    for (int c = 0; c < 4; ++c)
      #pragma unroll
      for (int r = 0; r < 8; ++r) acc[c][r] *= sc[r];

    asm volatile("s_wait_dscnt 0" ::: "memory");   // cross-lane LDS RAW fence

    // ---- read P back in A layout (16x32, contraction = key index) ----
    v16bf ap;
    {
      const float* pr = &lds[wv][m][0];
      #pragma unroll
      for (int i = 0; i < 8; ++i) {
        ap[i]     = (__bf16)pr[kh + i];
        ap[8 + i] = (__bf16)pr[kh + 8 + i];
      }
    }

    // ---- O += P @ V : batch the 4 V^T B-tile loads, then 4 WMMAs ----
    v16bf bv[4];
    #pragma unroll
    for (int c = 0; c < 4; ++c)
      bv[c] = *(const v16bf*)(Vtbh + (size_t)(c * 16 + m) * N_ + j + kh);
    #pragma unroll
    for (int c = 0; c < 4; ++c)
      acc[c] = __builtin_amdgcn_wmma_f32_16x16x32_bf16(false, ap, false, bv[c],
                                                       (short)0, acc[c], false, false);
  }

  // ---- normalize + residual ----
  float inv[8];
  #pragma unroll
  for (int r = 0; r < 8; ++r) inv[r] = 1.0f / lrow[r];
  const int bidx = bh >> 4, head = bh & 15;
  #pragma unroll
  for (int c = 0; c < 4; ++c) {
    #pragma unroll
    for (int r = 0; r < 8; ++r) {
      const int row = r + hi * 8;
      const size_t idx = ((size_t)bidx * N_ + qt * 16 + row) * D_
                       + head * HD_ + c * 16 + m;
      out[idx] = x[idx] + acc[c][r] * inv[r];
    }
  }
}

// ---------------------------------------------------------------------------
extern "C" void kernel_launch(void* const* d_in, const int* in_sizes, int n_in,
                              void* d_out, int out_size, void* d_ws, size_t ws_size,
                              hipStream_t stream)
{
  (void)in_sizes; (void)n_in; (void)out_size; (void)ws_size;
  const float* x  = (const float*)d_in[0];
  const float* Wq = (const float*)d_in[1];
  const float* bq = (const float*)d_in[2];
  const float* Wk = (const float*)d_in[3];
  const float* bk = (const float*)d_in[4];
  const float* Wv = (const float*)d_in[5];
  const float* bv = (const float*)d_in[6];
  float* out = (float*)d_out;

  // 0) X and W -> bf16 in ws: (8M + 3M) elems / 8 per thread / 256 per block
  cvt_bf16<<<5632, 256, 0, stream>>>(x, Wq, Wk, Wv, d_ws);
  // 1) 3 projections * 64 row-blocks * 16 col-strips = 3072 blocks
  qkv_gemm<<<3072, 256, 0, stream>>>(bq, bk, bv, d_ws);
  // 2) 64 (b,h) * 128 q-tiles = 8192 waves, 8 waves/block
  attn<<<1024, 256, 0, stream>>>(x, d_ws, out);
}
